// OutputPPBlock_9208409883359
// MI455X (gfx1250) — compile-verified, matrix operand order checked by reading, not run on hardware
//
#include <hip/hip_runtime.h>
#include <hip/hip_bf16.h>

#define N_NODES_C     25000
#define N_EDGES_C     800000
#define NUM_GRAPHS_C  512
#define EMB_C         128
#define OUT_EMB_C     256
#define NUM_RADIAL_C  6
#define NUM_DENSE_C   3
#define NUM_TARGETS_C 12
#define N_PAD_C       25008   // 25000 rounded up to multiple of 16

typedef __attribute__((ext_vector_type(16))) __bf16 v16bf;
typedef __attribute__((ext_vector_type(8)))  __bf16 v8bf;
typedef __attribute__((ext_vector_type(8)))  float  v8f;

// ---------------------------------------------------------------- zero fill
__global__ void zero_f32_kernel(float* __restrict__ p, int n) {
  int i = blockIdx.x * blockDim.x + threadIdx.x;
  if (i < n) p[i] = 0.0f;
}

// ------------------------------------------------------- f32 -> bf16 (flat)
__global__ void cvt_bf16_kernel(const float* __restrict__ src,
                                __bf16* __restrict__ dst, int n) {
  int i = blockIdx.x * blockDim.x + threadIdx.x;
  if (i < n) dst[i] = (__bf16)src[i];
}

// ------------------------------------------- weight f32 -> bf16, transposed
// W is [K x Nw] row-major. Bt is [Nw x K]: Bt[n*K+k].
__global__ void convert_T_bf16_kernel(const float* __restrict__ W,
                                      __bf16* __restrict__ Bt, int K, int Nw) {
  int i = blockIdx.x * blockDim.x + threadIdx.x;
  if (i < K * Nw) {
    int k = i / Nw;
    int n = i - k * Nw;
    Bt[n * K + k] = (__bf16)W[i];
  }
}

// --------------------------------------------- edge message + scatter to node
// one wave per edge; each lane handles 4 channels (float4).
__global__ void edge_scatter_kernel(const float* __restrict__ m,
                                    const float* __restrict__ rbf,
                                    const int*   __restrict__ edge_src,
                                    const float* __restrict__ w_rbf,
                                    float*       __restrict__ t_node) {
  long long i = (long long)blockIdx.x * blockDim.x + threadIdx.x;
  if (i >= (long long)N_EDGES_C * 32) return;
  int e = (int)(i >> 5);
  int q = ((int)i & 31) << 2;            // channel base: 0,4,...,124
  const float* re = rbf + e * NUM_RADIAL_C;
  float r0 = 0.f, r1 = 0.f, r2 = 0.f, r3 = 0.f;
#pragma unroll
  for (int k = 0; k < NUM_RADIAL_C; ++k) {
    float rk = re[k];
    const float4 w4 = *(const float4*)(w_rbf + k * EMB_C + q);
    r0 = fmaf(rk, w4.x, r0);
    r1 = fmaf(rk, w4.y, r1);
    r2 = fmaf(rk, w4.z, r2);
    r3 = fmaf(rk, w4.w, r3);
  }
  const float4 m4 = *(const float4*)(m + (long long)e * EMB_C + q);
  float* dst = t_node + (long long)edge_src[e] * EMB_C + q;
  atomicAdd(dst + 0, m4.x * r0);
  atomicAdd(dst + 1, m4.y * r1);
  atomicAdd(dst + 2, m4.z * r2);
  atomicAdd(dst + 3, m4.w * r3);
}

// ------------------------------------------------------- WMMA bf16 GEMM
// C[M x 256] = act( A[M x K] @ B[K x 256] + bias ), all bf16 operands,
// f32 accumulate, bf16 output. B supplied transposed as Bt[256 x K].
// One wave computes one 16-row tile x 4 consecutive 16-col tiles, so a single
// A fragment feeds 4 back-to-back WMMAs with independent accumulators.
// ISA 16-bit A layout (16x32 K-slab): lane = row (mod 16), kbase=(lane<16)?0:8,
// j=0..7 -> K=kbase+j ; j=8..15 -> K=kbase+16+(j-8). B symmetric with
// lane = column. C/D f32: lanes 0-15: N=lane, VGPR r -> M=r; lanes 16-31:
// N=lane-16, VGPR r -> M=8+r.
template <bool ACT>
__global__ void gemm_bf16_wmma_kernel(const __bf16* __restrict__ A,
                                      const __bf16* __restrict__ Bt,
                                      const float*  __restrict__ bias,
                                      __bf16*       __restrict__ C,
                                      int Mtiles, int K) {
  int wave = (int)((blockIdx.x * blockDim.x + threadIdx.x) >> 5);
  int lane = threadIdx.x & 31;
  if (wave >= Mtiles * 4) return;        // wave-uniform: EXEC all-ones below
  int tm  = wave >> 2;
  int tnb = (wave & 3) << 2;             // first of 4 N-tiles

  int l15   = lane & 15;
  int kbase = (lane >> 4) << 3;          // 0 or 8
  const __bf16* Arow = A + (size_t)(tm * 16 + l15) * K + kbase;
  const __bf16* B0 = Bt + (size_t)((tnb + 0) * 16 + l15) * K + kbase;
  const __bf16* B1 = Bt + (size_t)((tnb + 1) * 16 + l15) * K + kbase;
  const __bf16* B2 = Bt + (size_t)((tnb + 2) * 16 + l15) * K + kbase;
  const __bf16* B3 = Bt + (size_t)((tnb + 3) * 16 + l15) * K + kbase;

  v8f c0 = {}, c1 = {}, c2 = {}, c3 = {};
  for (int kk = 0; kk < K; kk += 32) {
    v8bf alo = *(const v8bf*)(Arow + kk);
    v8bf ahi = *(const v8bf*)(Arow + kk + 16);
    v16bf a;
#pragma unroll
    for (int j = 0; j < 8; ++j) { a[j] = alo[j]; a[8 + j] = ahi[j]; }

    v16bf b;
    v8bf blo, bhi;
    blo = *(const v8bf*)(B0 + kk); bhi = *(const v8bf*)(B0 + kk + 16);
#pragma unroll
    for (int j = 0; j < 8; ++j) { b[j] = blo[j]; b[8 + j] = bhi[j]; }
    c0 = __builtin_amdgcn_wmma_f32_16x16x32_bf16(false, a, false, b, (short)0, c0, false, false);

    blo = *(const v8bf*)(B1 + kk); bhi = *(const v8bf*)(B1 + kk + 16);
#pragma unroll
    for (int j = 0; j < 8; ++j) { b[j] = blo[j]; b[8 + j] = bhi[j]; }
    c1 = __builtin_amdgcn_wmma_f32_16x16x32_bf16(false, a, false, b, (short)0, c1, false, false);

    blo = *(const v8bf*)(B2 + kk); bhi = *(const v8bf*)(B2 + kk + 16);
#pragma unroll
    for (int j = 0; j < 8; ++j) { b[j] = blo[j]; b[8 + j] = bhi[j]; }
    c2 = __builtin_amdgcn_wmma_f32_16x16x32_bf16(false, a, false, b, (short)0, c2, false, false);

    blo = *(const v8bf*)(B3 + kk); bhi = *(const v8bf*)(B3 + kk + 16);
#pragma unroll
    for (int j = 0; j < 8; ++j) { b[j] = blo[j]; b[8 + j] = bhi[j]; }
    c3 = __builtin_amdgcn_wmma_f32_16x16x32_bf16(false, a, false, b, (short)0, c3, false, false);
  }

  int mbase = tm * 16 + ((lane >> 4) << 3);
  __bf16* Cm = C + (size_t)mbase * OUT_EMB_C;
#pragma unroll
  for (int t = 0; t < 4; ++t) {
    v8f c = (t == 0) ? c0 : (t == 1) ? c1 : (t == 2) ? c2 : c3;
    int n = (tnb + t) * 16 + l15;
    float bn = bias[n];
#pragma unroll
    for (int r = 0; r < 8; ++r) {
      float x = c[r] + bn;
      if (ACT) x = x * __builtin_amdgcn_rcpf(1.0f + __expf(-x));  // silu
      Cm[(size_t)r * OUT_EMB_C + n] = (__bf16)x;
    }
  }
}

// ------------------------------------- final 256->12 projection + graph sum
__global__ void final_readout_kernel(const __bf16* __restrict__ h,
                                     const float*  __restrict__ w_final,
                                     const int*    __restrict__ node2graph,
                                     float*        __restrict__ out) {
  int i = blockIdx.x * blockDim.x + threadIdx.x;
  if (i >= N_NODES_C * NUM_TARGETS_C) return;
  int node = i / NUM_TARGETS_C;
  int t    = i - node * NUM_TARGETS_C;
  const __bf16* hr = h + (size_t)node * OUT_EMB_C;
  float acc = 0.0f;
#pragma unroll 8
  for (int k = 0; k < OUT_EMB_C; ++k)
    acc = fmaf((float)hr[k], w_final[k * NUM_TARGETS_C + t], acc);
  atomicAdd(&out[node2graph[node] * NUM_TARGETS_C + t], acc);
}

// ---------------------------------------------------------------------------
extern "C" void kernel_launch(void* const* d_in, const int* in_sizes, int n_in,
                              void* d_out, int out_size, void* d_ws, size_t ws_size,
                              hipStream_t stream) {
  const float* m          = (const float*)d_in[0];
  const float* rbf        = (const float*)d_in[1];
  const int*   edge_src   = (const int*)d_in[2];
  const int*   node2graph = (const int*)d_in[3];
  const float* w_rbf      = (const float*)d_in[4];
  const float* w_up       = (const float*)d_in[5];
  const float* w_dense    = (const float*)d_in[6];
  const float* b_dense    = (const float*)d_in[7];
  const float* w_final    = (const float*)d_in[8];
  float* out = (float*)d_out;

  // workspace layout
  char* ws = (char*)d_ws;
  size_t o = 0;
  auto take = [&](size_t bytes) -> void* {
    void* p = ws + o;
    o += (bytes + 255) & ~(size_t)255;
    return p;
  };
  float*  t_node = (float*) take(sizeof(float)  * (size_t)N_PAD_C * EMB_C);
  __bf16* a0     = (__bf16*)take(sizeof(__bf16) * (size_t)N_PAD_C * EMB_C);
  __bf16* a1     = (__bf16*)take(sizeof(__bf16) * (size_t)N_PAD_C * OUT_EMB_C);
  __bf16* a2     = (__bf16*)take(sizeof(__bf16) * (size_t)N_PAD_C * OUT_EMB_C);
  __bf16* wupT   = (__bf16*)take(sizeof(__bf16) * (size_t)OUT_EMB_C * EMB_C);
  __bf16* wdT    = (__bf16*)take(sizeof(__bf16) * (size_t)NUM_DENSE_C * OUT_EMB_C * OUT_EMB_C);
  float*  zbias  = (float*) take(sizeof(float)  * OUT_EMB_C);

  const int TB = 256;

  // 1) zero node accumulator (covers padded rows), zero-bias, output buffer
  {
    int n = N_PAD_C * EMB_C;
    zero_f32_kernel<<<(n + TB - 1) / TB, TB, 0, stream>>>(t_node, n);
    zero_f32_kernel<<<1, TB, 0, stream>>>(zbias, OUT_EMB_C);
    int ng = NUM_GRAPHS_C * NUM_TARGETS_C;
    zero_f32_kernel<<<(ng + TB - 1) / TB, TB, 0, stream>>>(out, ng);
  }

  // 2) weights -> transposed bf16
  {
    int n = EMB_C * OUT_EMB_C;
    convert_T_bf16_kernel<<<(n + TB - 1) / TB, TB, 0, stream>>>(w_up, wupT, EMB_C, OUT_EMB_C);
    int nd = OUT_EMB_C * OUT_EMB_C;
    for (int l = 0; l < NUM_DENSE_C; ++l)
      convert_T_bf16_kernel<<<(nd + TB - 1) / TB, TB, 0, stream>>>(
          w_dense + (size_t)l * nd, wdT + (size_t)l * nd, OUT_EMB_C, OUT_EMB_C);
  }

  // 3) edge messages + scatter-sum into nodes (one wave per edge)
  {
    long long n = (long long)N_EDGES_C * 32;
    edge_scatter_kernel<<<(unsigned)((n + TB - 1) / TB), TB, 0, stream>>>(
        m, rbf, edge_src, w_rbf, t_node);
  }

  // 4) node accumulator -> bf16 activations
  {
    int n = N_PAD_C * EMB_C;
    cvt_bf16_kernel<<<(n + TB - 1) / TB, TB, 0, stream>>>(t_node, a0, n);
  }

  // 5) MLP chain via WMMA: up-proj (no act), 3x dense(silu). Nw fixed at 256.
  const int Mtiles = N_PAD_C / 16;
  const int waves  = Mtiles * 4;                 // 4 N-tiles per wave
  const unsigned gblocks = (unsigned)(((long long)waves * 32 + TB - 1) / TB);
  const int nd = OUT_EMB_C * OUT_EMB_C;
  gemm_bf16_wmma_kernel<false><<<gblocks, TB, 0, stream>>>(a0, wupT,          zbias,                 a1, Mtiles, EMB_C);
  gemm_bf16_wmma_kernel<true ><<<gblocks, TB, 0, stream>>>(a1, wdT + 0 * nd,  b_dense + 0*OUT_EMB_C, a2, Mtiles, OUT_EMB_C);
  gemm_bf16_wmma_kernel<true ><<<gblocks, TB, 0, stream>>>(a2, wdT + 1 * nd,  b_dense + 1*OUT_EMB_C, a1, Mtiles, OUT_EMB_C);
  gemm_bf16_wmma_kernel<true ><<<gblocks, TB, 0, stream>>>(a1, wdT + 2 * nd,  b_dense + 2*OUT_EMB_C, a2, Mtiles, OUT_EMB_C);

  // 6) final projection + per-graph readout
  {
    int n = N_NODES_C * NUM_TARGETS_C;
    final_readout_kernel<<<(n + TB - 1) / TB, TB, 0, stream>>>(a2, w_final, node2graph, out);
  }
}
